// ScaledDotProductAttention_65876208386478
// MI455X (gfx1250) — matrix operand order repaired and checked by reading, not compiled, not run on hardware
//
#include <hip/hip_runtime.h>
#include <hip/hip_bf16.h>
#include <math.h>

// ---------------------------------------------------------------------------
// Fused attention for B=16,H=16,N=512,D=64 (fp32 in, fp32 out).
// outputs (tuple order): attn [B,H,N,N] then out [B,H,N,D], concatenated.
// Memory-bound (~644 MB moved -> ~28us floor @ 23.3 TB/s).
//  - matmuls: v_wmma_f32_16x16x32_f16 (f16 operands, fp32 accumulate)
//  - bias tile (the largest input stream, 256 MB total) is copied into LDS
//    with global_load_async_to_lds_b128 (ASYNCcnt) overlapping the QK^T WMMAs
// ---------------------------------------------------------------------------

typedef __attribute__((ext_vector_type(16))) _Float16 v16h;
typedef __attribute__((ext_vector_type(8)))  float    v8f;

#define ATTN_B 16
#define ATTN_H 16
#define ATTN_N 512
#define ATTN_D 64
#define ATTN_NEG (-1000000000.0f)

// --- WMMA wrapper: D = A(16x32 f16) * B(32x16 f16) + C(16x16 f32) ----------
__device__ __forceinline__ v8f wmma_f32_f16(v16h a, v16h b, v8f c) {
    return __builtin_amdgcn_wmma_f32_16x16x32_f16(
        /*neg_a=*/false, a, /*neg_b=*/false, b,
        /*c_mod=*/(short)0, c, /*reuse_a=*/false, /*reuse_b=*/false);
}

// --- A-matrix (16x32, f16) load from row-major fp32 rows -------------------
// ISA layout: lane L<16 -> row M=L, K = {kb+0..7, kb+16..23};
//             lane L>=16 -> row M=L-16, K = {kb+8..15, kb+24..31}.
__device__ __forceinline__ v16h load_a_rows_f32(const float* p, int ldr,
                                                int lane, int kb) {
    const int r  = lane & 15;
    const int hf = lane >> 4;
    const float* q = p + r * ldr + kb + hf * 8;
    v16h a;
#pragma unroll
    for (int i = 0; i < 8; ++i) a[i] = (_Float16)q[i];
#pragma unroll
    for (int i = 0; i < 8; ++i) a[8 + i] = (_Float16)q[16 + i];
    return a;
}

// --- B-matrix (32x16, f16) for Q*K^T: B(d,n) = K[cb+n][db+d] ---------------
// ISA layout: lane L holds column N=L%16; v16h = K {8h+0..7, 16+8h+0..7}, h=L/16.
__device__ __forceinline__ v16h load_b_ktrans(const float* kmat, int cb,
                                              int db, int lane) {
    v16h b;
#pragma unroll
    for (int n = 0; n < 16; ++n)
        b[n] = (_Float16)kmat[(size_t)(cb + (lane & 15)) * ATTN_D + db + (lane >> 4) * 8 + (n & 7) + (n >> 3) * 16];
    return b;
}

// --- B-matrix (32x16, f16) for attn*V: B(k,n) = V[kb+k][db+n] --------------
// lane L holds column N=L%16; v16h = K {8h+0..7, 16+8h+0..7}, h=L/16.
__device__ __forceinline__ v16h load_b_vrows(const float* vmat, int kb,
                                             int db, int lane) {
    const float* p = vmat + (size_t)(kb + (lane >> 4) * 8) * ATTN_D + db + (lane & 15);
    v16h b;
#pragma unroll
    for (int n = 0; n < 16; ++n) b[n] = (_Float16)p[((n & 7) + (n >> 3) * 16) * ATTN_D];
    return b;
}

// --- A-matrix (16x32, f16) from LDS score tile, normalized by inv ----------
__device__ __forceinline__ v16h load_a_scores(const float (*sS)[ATTN_N],
                                              int kb, int lane, float inv) {
    const int r  = lane & 15;
    const int hf = lane >> 4;
    const float* q = &sS[r][kb + hf * 8];
    v16h a;
#pragma unroll
    for (int i = 0; i < 8; ++i) a[i] = (_Float16)(q[i] * inv);
#pragma unroll
    for (int i = 0; i < 8; ++i) a[8 + i] = (_Float16)(q[16 + i] * inv);
    return a;
}

__global__ __launch_bounds__(256)
void sdpa_wmma_kernel(const float* __restrict__ Q,
                      const float* __restrict__ K,
                      const float* __restrict__ V,
                      const unsigned char* __restrict__ mask,  // bool [B,1,N,N]
                      const float* __restrict__ bias,
                      float* __restrict__ attn,
                      float* __restrict__ out) {
    __shared__ float sS[16][ATTN_N];      // 32 KB: bias tile -> score tile
    __shared__ float sRed[256];           // reduction scratch
    __shared__ float sMax[16];
    __shared__ float sSum[16];
    __shared__ float sAcc[4][16][16];     // attn@V partial accumulators

    const int tid  = threadIdx.x;
    const int lane = tid & 31;
    const int wv   = tid >> 5;            // 0..7 (wave32)
    const int r    = lane & 15;
    const int hf   = lane >> 4;

    const int blk = blockIdx.x;           // B*H*(N/16) = 8192
    const int qt  = blk & 31;             // q tile within head
    const int bh  = blk >> 5;             // fused (b,h)
    const int b   = bh >> 4;              // H = 16
    const int q0  = qt * 16;

    const float* Qp = Q    + ((size_t)bh * ATTN_N + q0) * ATTN_D;
    const float* Kp = K    + (size_t)bh * ATTN_N * ATTN_D;
    const float* Vp = V    + (size_t)bh * ATTN_N * ATTN_D;
    const float* Bp = bias + (size_t)bh * ATTN_N * ATTN_N + (size_t)q0 * ATTN_N;
    const unsigned char* Mp = mask + (size_t)b * ATTN_N * ATTN_N + (size_t)q0 * ATTN_N;
    float* Ap = attn + (size_t)bh * ATTN_N * ATTN_N + (size_t)q0 * ATTN_N;
    float* Op = out  + ((size_t)bh * ATTN_N + q0) * ATTN_D;

    // ---- Async copy of the 16x512 fp32 bias tile (contiguous 32 KB in both
    // global memory and sS) into LDS. 8 x b128 per thread, fully coalesced
    // (within an instruction, lanes cover 512B contiguous). Runs on the
    // async path (ASYNCcnt) in the background of Q/K loads and QK^T WMMAs.
    {
        const unsigned lds_base = (unsigned)(uintptr_t)(&sS[0][0]);
#pragma unroll
        for (int j = 0; j < 8; ++j) {
            const unsigned elt = (unsigned)(j * 256 + tid);   // float4 index
            const unsigned la  = lds_base + elt * 16u;
            const unsigned long long ga =
                (unsigned long long)(uintptr_t)(Bp + (size_t)elt * 4);
            asm volatile("global_load_async_to_lds_b128 %0, %1, off"
                         :: "v"(la), "v"(ga) : "memory");
        }
    }

    // Warm caches for V (phase 4) and mask (phase 1 epilogue).
    __builtin_prefetch(Vp + (size_t)tid * 128, 0, 0);
    __builtin_prefetch(Mp + (size_t)tid * 32, 0, 0);

    // ---------------- Phase 1a: QK^T WMMAs (K=64 -> 2 chunks of 32) --------
    const v16h aq0 = load_a_rows_f32(Qp, ATTN_D, lane, 0);
    const v16h aq1 = load_a_rows_f32(Qp, ATTN_D, lane, 32);

    v8f acc[4];
#pragma unroll
    for (int t = 0; t < 4; ++t) {
        const int cb = (wv * 4 + t) * 16;       // column tile base
        v8f c = {};
        const v16h b0 = load_b_ktrans(Kp, cb, 0, lane);
        c = wmma_f32_f16(aq0, b0, c);
        const v16h b1 = load_b_ktrans(Kp, cb, 32, lane);
        c = wmma_f32_f16(aq1, b1, c);
        acc[t] = c;
    }

    // Bias tile must be fully resident in LDS before the epilogue.
    asm volatile("s_wait_asynccnt 0x0" ::: "memory");
    __syncthreads();

    // ---------------- Phase 1b: scale + bias + mask + fixups ---------------
#pragma unroll
    for (int t = 0; t < 4; ++t) {
        const int cb = (wv * 4 + t) * 16;
#pragma unroll
        for (int v = 0; v < 8; ++v) {
            const int m  = v + hf * 8;          // local q row (C/D layout)
            const int kg = cb + r;              // key column
            const int qg = q0 + m;              // global q row in this head
            float s = acc[t][v] * 0.125f + sS[m][kg];   // sS holds bias
            if (!Mp[m * ATTN_N + kg]) s = ATTN_NEG;
            if (qg == 0 && kg >= 2)   s = ATTN_NEG;
            if (kg == 0 && qg >= 2)   s = ATTN_NEG;
            sS[m][kg] = s;
        }
    }
    __syncthreads();

    // ---------------- Phase 2: softmax over k (fp32 in LDS) ----------------
    const int rr  = tid >> 4;                   // row 0..15
    const int seg = tid & 15;                   // 16 threads per row
    float mx = -3.0e38f;
#pragma unroll 8
    for (int j = 0; j < 32; ++j) mx = fmaxf(mx, sS[rr][seg + j * 16]);
    sRed[tid] = mx;
    __syncthreads();
    if (seg == 0) {
        float m2 = sRed[tid];
#pragma unroll
        for (int k = 1; k < 16; ++k) m2 = fmaxf(m2, sRed[tid + k]);
        sMax[rr] = m2;
    }
    __syncthreads();

    const float rmax = sMax[rr];
    float sm = 0.0f;
#pragma unroll 8
    for (int j = 0; j < 32; ++j) {
        const int cc = seg + j * 16;
        const float e = __expf(sS[rr][cc] - rmax);
        sS[rr][cc] = e;
        sm += e;
    }
    sRed[tid] = sm;
    __syncthreads();
    if (seg == 0) {
        float s2 = sRed[tid];
#pragma unroll
        for (int k = 1; k < 16; ++k) s2 += sRed[tid + k];
        sSum[rr] = s2;
    }
    __syncthreads();

    // ---------------- Phase 3: write attn tile (float4 stores) -------------
    {
        const float invr = 1.0f / sSum[rr];
        const int colb = seg * 32;              // 32 floats / thread
        float* dst = Ap + (size_t)rr * ATTN_N + colb;
#pragma unroll
        for (int j = 0; j < 32; j += 4) {
            float4 vv;
            vv.x = sS[rr][colb + j + 0] * invr;
            vv.y = sS[rr][colb + j + 1] * invr;
            vv.z = sS[rr][colb + j + 2] * invr;
            vv.w = sS[rr][colb + j + 3] * invr;
            *(float4*)(dst + j) = vv;
        }
    }

    // ---------------- Phase 4: out = attn @ V (split-K over 8 waves) -------
    const int dt = wv & 3;                      // d tile: 4 x 16 = 64
    const int kh = wv >> 2;                     // k half: 0 or 1
    const int db = dt * 16;
    const float invA = 1.0f / sSum[r];          // per-lane row normalizer
    v8f o = {};
#pragma unroll
    for (int ch = 0; ch < 8; ++ch) {
        const int kb = kh * 256 + ch * 32;
        const v16h aa = load_a_scores(sS, kb, lane, invA);
        const v16h bb = load_b_vrows(Vp, kb, db, lane);
        o = wmma_f32_f16(aa, bb, o);
    }
    if (kh == 1) {
#pragma unroll
        for (int v = 0; v < 8; ++v) sAcc[dt][v + hf * 8][r] = o[v];
    }
    __syncthreads();
    if (kh == 0) {
#pragma unroll
        for (int v = 0; v < 8; ++v) {
            const int m = v + hf * 8;
            Op[m * ATTN_D + db + r] = o[v] + sAcc[dt][m][r];
        }
    }
}

extern "C" void kernel_launch(void* const* d_in, const int* in_sizes, int n_in,
                              void* d_out, int out_size, void* d_ws, size_t ws_size,
                              hipStream_t stream) {
    // setup_inputs order: Q, K, V, mask(bool), bias
    const float* Q = (const float*)d_in[0];
    const float* K = (const float*)d_in[1];
    const float* V = (const float*)d_in[2];
    const unsigned char* mask = (const unsigned char*)d_in[3];  // jnp bool = 1 byte
    const float* bias = (const float*)d_in[4];

    float* attn = (float*)d_out;                                       // [B,H,N,N]
    float* out  = attn + (size_t)ATTN_B * ATTN_H * ATTN_N * ATTN_N;    // [B,H,N,D]

    const int blocks = ATTN_B * ATTN_H * (ATTN_N / 16);  // 8192
    sdpa_wmma_kernel<<<blocks, 256, 0, stream>>>(Q, K, V, mask, bias, attn, out);
}
